// LocallyGroupedAttn_20040317403633
// MI455X (gfx1250) — matrix operand order repaired and checked by reading, not compiled, not run on hardware
//
#include <hip/hip_runtime.h>
#include <hip/hip_bf16.h>

typedef __attribute__((ext_vector_type(16))) _Float16 v16h;
typedef __attribute__((ext_vector_type(8)))  _Float16 v8h;
typedef __attribute__((ext_vector_type(8)))  float    v8f;

#define C_DIM 256
#define L_WIN 64
#define GRID_W 240

static __device__ __forceinline__ int lane_id() { return threadIdx.x & 31; }

static __device__ __forceinline__ v8f vzero8() {
  v8f z;
#pragma unroll
  for (int i = 0; i < 8; ++i) z[i] = 0.0f;
  return z;
}

// A-fragment (16x32 f16) from row-major LDS, ld in halfs.
// Contiguous-k order (same as packed B): lane l -> M = l&15, Kbase = (l>>4)*16 ; halfs 0..7 = K base..base+7,
// halfs 8..15 = K base+8..base+15  -> two 16B ds loads.
static __device__ __forceinline__ v16h load_a_frag(const _Float16* A, int ld) {
  const int l = lane_id();
  const _Float16* p = A + (l & 15) * ld + ((l >> 4) << 4);
  v8h lo = *(const v8h*)(p);
  v8h hi = *(const v8h*)(p + 8);
  v16h r;
#pragma unroll
  for (int i = 0; i < 8; ++i) { r[i] = lo[i]; r[i + 8] = hi[i]; }
  return r;
}

// B-fragment (32x16 f16) from pre-swizzled fragment storage: lane-major 16 halfs.
static __device__ __forceinline__ v16h load_b_frag(const _Float16* F) {
  return *(const v16h*)(F + lane_id() * 16);
}

static __device__ __forceinline__ void prefetch_b_frag(const _Float16* F) {
  __builtin_prefetch(F + lane_id() * 16, 0, 2);   // -> global_prefetch_b8
}

static __device__ __forceinline__ v8f wmma_f16(v16h a, v16h b, v8f c) {
  return __builtin_amdgcn_wmma_f32_16x16x32_f16(false, a, false, b, (short)0, c,
                                                false, false);
}

// C-frag (16x16 f32): element r -> (M = r + 8*(l>>4), N = l&15)
static __device__ __forceinline__ void store_c_rm_h(_Float16* D, int ld, v8f c) {
  const int l = lane_id();
  const int n = l & 15, mb = (l >> 4) * 8;
#pragma unroll
  for (int r = 0; r < 8; ++r) D[(mb + r) * ld + n] = (_Float16)c[r];
}

static __device__ __forceinline__ void store_c_rm_f(float* D, int ld, v8f c) {
  const int l = lane_id();
  const int n = l & 15, mb = (l >> 4) * 8;
#pragma unroll
  for (int r = 0; r < 8; ++r) D[(mb + r) * ld + n] = c[r];
}

// transposed store: element (M,N) -> D[N*ld + M]; per lane contiguous 8 halfs.
static __device__ __forceinline__ void store_c_tr_h(_Float16* D, int ld, v8f c) {
  const int l = lane_id();
  const int n = l & 15, mb = (l >> 4) * 8;
  v8h h;
#pragma unroll
  for (int r = 0; r < 8; ++r) h[r] = (_Float16)c[r];
  *(v8h*)(D + n * ld + mb) = h;
}

// Store C-frag (tile mt of a 32-row K block) directly into B-fragment layout:
// element (k = 16*mt + 8*hi + r, n): lane lb = (mt&1)*16 + (n&15), half j = k&15.
// Per lane this is 8 consecutive halfs -> one 16B ds store.
static __device__ __forceinline__ void store_c_as_bfrag(_Float16* F, v8f c, int mt) {
  const int l = lane_id();
  const int lb = ((mt & 1) << 4) + (l & 15);
  const int hi = l >> 4;
  v8h h;
#pragma unroll
  for (int r = 0; r < 8; ++r) h[r] = (_Float16)c[r];
  *(v8h*)(F + lb * 16 + hi * 8) = h;
}

// ---------------------------------------------------------------------------
// Weight pre-pack: f32 [K][N] -> f16 fragment-major:
// frag = kt2*(N/16)+nt ; within frag: lane*16 + j ;
// value = W[kt2*32+(lane>>4)*16+j][nt*16+(lane&15)]
// ---------------------------------------------------------------------------
__global__ void pack_w_kernel(const float* __restrict__ W, _Float16* __restrict__ out,
                              int K, int N) {
  int idx = blockIdx.x * blockDim.x + threadIdx.x;
  if (idx >= K * N) return;
  int j    = idx & 15;
  int lane = (idx >> 4) & 31;
  int frag = idx >> 9;
  int NT   = N >> 4;
  int kt2  = frag / NT;
  int nt   = frag - kt2 * NT;
  int k = kt2 * 32 + (lane >> 4) * 16 + j;
  int n = nt * 16 + (lane & 15);
  out[idx] = (_Float16)W[k * N + n];
}

// ---------------------------------------------------------------------------
// Main fused LoFTR encoder-layer kernel: 1 block = 1 window (8x8 tokens, C=256)
// ---------------------------------------------------------------------------
#define SMEM_BYTES (65536 /*cat*/ + 65536 /*R1*/ + 32768 /*Vfr*/ + 16384 /*KVfr*/ \
                    + 65536 /*scr*/ + 1024 /*Ksum*/ + 2048 /*Zs*/)

__global__ __launch_bounds__(256)
void loftr_layer_kernel(const float* __restrict__ x,
                        const _Float16* __restrict__ Wq, const _Float16* __restrict__ Wk,
                        const _Float16* __restrict__ Wv, const _Float16* __restrict__ Wm,
                        const _Float16* __restrict__ W1, const _Float16* __restrict__ W2,
                        const float* __restrict__ g1, const float* __restrict__ b1,
                        const float* __restrict__ g2, const float* __restrict__ b2,
                        float* __restrict__ out) {
  extern __shared__ char smem[];
  _Float16* cat  = (_Float16*)(smem);                    // [64][512] f16 (x | msg)
  _Float16* R1   = (_Float16*)(smem + 65536);            // 64KB union region
  _Float16* Qh   = R1;                                   // [64][256] f16
  _Float16* Kt   = R1 + 64 * 256;                        // [256][64] f16 (K^T)
  _Float16* hA   = R1;                                   // [64][512] f16 (MLP hidden)
  _Float16* Vfr  = (_Float16*)(smem + 131072);           // 32 B-frags (V) / msgA[64][256]
  _Float16* msgA = Vfr;
  _Float16* KVfr = (_Float16*)(smem + 131072 + 32768);   // 16 B-frags (KV)
  float*    scr  = (float*)(smem + 131072 + 32768 + 16384);            // [64][256] f32
  float*    Ksum = (float*)(smem + 131072 + 32768 + 16384 + 65536);    // [256]
  float*    Zs   = Ksum + 256;                                         // [64*8]

  const int tid  = threadIdx.x;
  const int wave = tid >> 5;
  const int wblk = blockIdx.x;            // 0..3599
  const int bb   = wblk / 900;
  const int rem  = wblk - bb * 900;
  const int hy   = rem / 30;
  const int wx   = rem - hy * 30;
  const long xbase = (long)bb * (GRID_W * GRID_W) * C_DIM;

  // ---- Phase 0: gather x window -> cat[:,0:256] as f16 --------------------
#pragma unroll
  for (int i = 0; i < 16; ++i) {
    int idx = tid + 256 * i;              // [64 tokens][64 float4]
    int t   = idx >> 6;
    int c4  = idx & 63;
    int n   = (hy * 8 + (t >> 3)) * GRID_W + wx * 8 + (t & 7);
    const float4 v = *(const float4*)(x + xbase + (long)n * C_DIM + c4 * 4);
    _Float16* d = cat + t * 512 + c4 * 4;
    d[0] = (_Float16)v.x; d[1] = (_Float16)v.y;
    d[2] = (_Float16)v.z; d[3] = (_Float16)v.w;
  }
  __syncthreads();

  // ---- Phase 1: q,k,v projections; elu+1; V scaled 1/L --------------------
  for (int ntl = 0; ntl < 2; ++ntl) {
    const int nt = wave * 2 + ntl;        // 0..15 column tiles
    v8f aq[4], ak[4], av[4];
#pragma unroll
    for (int m = 0; m < 4; ++m) { aq[m] = vzero8(); ak[m] = vzero8(); av[m] = vzero8(); }
    for (int kt = 0; kt < 8; ++kt) {
      v16h bq = load_b_frag(Wq + (kt * 16 + nt) * 512);
      v16h bk = load_b_frag(Wk + (kt * 16 + nt) * 512);
      v16h bv = load_b_frag(Wv + (kt * 16 + nt) * 512);
      prefetch_b_frag(Wq + (((kt + 1) & 7) * 16 + nt) * 512);
#pragma unroll
      for (int m = 0; m < 4; ++m) {
        v16h a = load_a_frag(cat + m * 16 * 512 + kt * 32, 512);
        aq[m] = wmma_f16(a, bq, aq[m]);
        ak[m] = wmma_f16(a, bk, ak[m]);
        av[m] = wmma_f16(a, bv, av[m]);
      }
    }
#pragma unroll
    for (int m = 0; m < 4; ++m) {
      v8f q = aq[m], k = ak[m], vv = av[m];
#pragma unroll
      for (int i = 0; i < 8; ++i) {
        q[i]  = q[i] > 0.0f ? q[i] + 1.0f : __expf(q[i]);   // elu(x)+1
        k[i]  = k[i] > 0.0f ? k[i] + 1.0f : __expf(k[i]);
        vv[i] = vv[i] * (1.0f / 64.0f);                     // v / L
      }
      store_c_rm_h(Qh + m * 16 * 256 + nt * 16, 256, q);
      store_c_tr_h(Kt + nt * 16 * 64 + m * 16, 64, k);      // K^T
      store_c_as_bfrag(Vfr + ((m >> 1) * 16 + nt) * 512, vv, m);
    }
  }
  __syncthreads();

  // ---- Phase 2a: Ksum[chan] = sum_s K[s][chan] (vectorized) ---------------
  {
    const v8h* kr = (const v8h*)(Kt + tid * 64);
    float s = 0.0f;
#pragma unroll
    for (int c8 = 0; c8 < 8; ++c8) {
      v8h h = kr[c8];
#pragma unroll
      for (int i = 0; i < 8; ++i) s += (float)h[i];
    }
    Ksum[tid] = s;
  }
  // ---- Phase 2b: KV[h] = K^T @ V per head (wave == head) ------------------
  {
    const int h = wave;
    for (int ntl = 0; ntl < 2; ++ntl) {
      v8f acc[2];
      acc[0] = vzero8(); acc[1] = vzero8();
      for (int kt2 = 0; kt2 < 2; ++kt2) {
        v16h b = load_b_frag(Vfr + (kt2 * 16 + 2 * h + ntl) * 512);
#pragma unroll
        for (int m = 0; m < 2; ++m) {
          v16h a = load_a_frag(Kt + (32 * h + m * 16) * 64 + kt2 * 32, 64);
          acc[m] = wmma_f16(a, b, acc[m]);
        }
      }
#pragma unroll
      for (int m = 0; m < 2; ++m)
        store_c_as_bfrag(KVfr + (h * 2 + ntl) * 512, acc[m], m);
    }
  }
  __syncthreads();

  // ---- Phase 2c: Z[l][h] = L / (Q[l,h,:].Ksum[h,:] + eps) -----------------
#pragma unroll
  for (int rep = 0; rep < 2; ++rep) {
    int i = tid + rep * 256;              // 0..511 = l*8+h
    int l = i >> 3, h = i & 7;
    const v8h*   qr  = (const v8h*)(Qh + l * 256 + h * 32);
    const float4* ks4 = (const float4*)(Ksum + h * 32);
    float dot = 0.0f;
#pragma unroll
    for (int c8 = 0; c8 < 4; ++c8) {
      v8h hv = qr[c8];
      float4 ka = ks4[c8 * 2], kb = ks4[c8 * 2 + 1];
      dot += (float)hv[0] * ka.x + (float)hv[1] * ka.y +
             (float)hv[2] * ka.z + (float)hv[3] * ka.w +
             (float)hv[4] * kb.x + (float)hv[5] * kb.y +
             (float)hv[6] * kb.z + (float)hv[7] * kb.w;
    }
    Zs[l * 8 + h] = 64.0f / (dot + 1e-6f); // fold *L into Z
  }
  __syncthreads();

  // ---- Phase 3: msg = (Q @ KV) * Z*L  -> msgA f16 (A-frag shared) ---------
  {
    v16h b0 = load_b_frag(KVfr + (wave * 2 + 0) * 512);
    v16h b1 = load_b_frag(KVfr + (wave * 2 + 1) * 512);
    const int hi = lane_id() >> 4;
#pragma unroll
    for (int m = 0; m < 4; ++m) {
      v16h a = load_a_frag(Qh + m * 16 * 256 + wave * 32, 256);  // K=32
      v8f c0 = wmma_f16(a, b0, vzero8());
      v8f c1 = wmma_f16(a, b1, vzero8());
#pragma unroll
      for (int r = 0; r < 8; ++r) {
        int row = m * 16 + hi * 8 + r;
        float z = Zs[row * 8 + wave];
        c0[r] *= z;
        c1[r] *= z;
      }
      store_c_rm_h(msgA + m * 16 * 256 + (wave * 2 + 0) * 16, 256, c0);
      store_c_rm_h(msgA + m * 16 * 256 + (wave * 2 + 1) * 16, 256, c1);
    }
  }
  __syncthreads();

  // ---- Phase 4: msg @ Wm -> scr f32 (2 column tiles / pass) ---------------
  {
    const int nt0 = wave * 2;
    v8f acc[2][4];
#pragma unroll
    for (int m = 0; m < 4; ++m) { acc[0][m] = vzero8(); acc[1][m] = vzero8(); }
    for (int kt = 0; kt < 8; ++kt) {
      v16h b0 = load_b_frag(Wm + (kt * 16 + nt0 + 0) * 512);
      v16h b1 = load_b_frag(Wm + (kt * 16 + nt0 + 1) * 512);
      prefetch_b_frag(Wm + (((kt + 1) & 7) * 16 + nt0) * 512);
#pragma unroll
      for (int m = 0; m < 4; ++m) {
        v16h a = load_a_frag(msgA + m * 16 * 256 + kt * 32, 256);
        acc[0][m] = wmma_f16(a, b0, acc[0][m]);
        acc[1][m] = wmma_f16(a, b1, acc[1][m]);
      }
    }
#pragma unroll
    for (int m = 0; m < 4; ++m) {
      store_c_rm_f(scr + m * 16 * 256 + (nt0 + 0) * 16, 256, acc[0][m]);
      store_c_rm_f(scr + m * 16 * 256 + (nt0 + 1) * 16, 256, acc[1][m]);
    }
  }
  __syncthreads();

  // ---- Phase 5: LN1 -> cat[:,256:512] f16 (4 threads / token) -------------
  {
    const int t = tid >> 2, qd = tid & 3;
    const float4* r4 = (const float4*)(scr + t * 256 + qd * 64);
    float s = 0.0f, s2 = 0.0f;
#pragma unroll
    for (int i = 0; i < 16; ++i) {
      float4 v = r4[i];
      s  += v.x + v.y + v.z + v.w;
      s2 += v.x * v.x + v.y * v.y + v.z * v.z + v.w * v.w;
    }
    s  += __shfl_xor(s, 1);  s  += __shfl_xor(s, 2);
    s2 += __shfl_xor(s2, 1); s2 += __shfl_xor(s2, 2);
    float mean = s * (1.0f / 256.0f);
    float var  = s2 * (1.0f / 256.0f) - mean * mean;
    float inv  = rsqrtf(var + 1e-5f);
    _Float16* d = cat + t * 512 + 256 + qd * 64;
    const float4* gg4 = (const float4*)(g1 + qd * 64);
    const float4* bb4 = (const float4*)(b1 + qd * 64);
#pragma unroll
    for (int i = 0; i < 16; ++i) {
      float4 v = r4[i], g = gg4[i], b = bb4[i];
      d[i * 4 + 0] = (_Float16)((v.x - mean) * inv * g.x + b.x);
      d[i * 4 + 1] = (_Float16)((v.y - mean) * inv * g.y + b.y);
      d[i * 4 + 2] = (_Float16)((v.z - mean) * inv * g.z + b.z);
      d[i * 4 + 3] = (_Float16)((v.w - mean) * inv * g.w + b.w);
    }
  }
  __syncthreads();

  // ---- Phase 6: h = relu(cat @ Wmlp1) -> hA f16 [64][512] -----------------
  for (int pp = 0; pp < 2; ++pp) {
    const int nt0 = wave * 4 + pp * 2;    // column tile pair
    v8f acc[2][4];
#pragma unroll
    for (int m = 0; m < 4; ++m) { acc[0][m] = vzero8(); acc[1][m] = vzero8(); }
    for (int kt = 0; kt < 16; ++kt) {
      v16h b0 = load_b_frag(W1 + (kt * 32 + nt0 + 0) * 512);
      v16h b1 = load_b_frag(W1 + (kt * 32 + nt0 + 1) * 512);
      prefetch_b_frag(W1 + (((kt + 1) & 15) * 32 + nt0) * 512);
#pragma unroll
      for (int m = 0; m < 4; ++m) {
        v16h a = load_a_frag(cat + m * 16 * 512 + kt * 32, 512);
        acc[0][m] = wmma_f16(a, b0, acc[0][m]);
        acc[1][m] = wmma_f16(a, b1, acc[1][m]);
      }
    }
#pragma unroll
    for (int m = 0; m < 4; ++m) {
#pragma unroll
      for (int p = 0; p < 2; ++p) {
        v8f c = acc[p][m];
#pragma unroll
        for (int i = 0; i < 8; ++i) c[i] = c[i] > 0.0f ? c[i] : 0.0f;
        store_c_rm_h(hA + m * 16 * 512 + (nt0 + p) * 16, 512, c);
      }
    }
  }
  __syncthreads();

  // ---- Phase 7: h2 = hA @ Wmlp2 -> scr f32 (2 column tiles / pass) --------
  {
    const int nt0 = wave * 2;
    v8f acc[2][4];
#pragma unroll
    for (int m = 0; m < 4; ++m) { acc[0][m] = vzero8(); acc[1][m] = vzero8(); }
    for (int kt = 0; kt < 16; ++kt) {
      v16h b0 = load_b_frag(W2 + (kt * 16 + nt0 + 0) * 512);
      v16h b1 = load_b_frag(W2 + (kt * 16 + nt0 + 1) * 512);
      prefetch_b_frag(W2 + (((kt + 1) & 15) * 16 + nt0) * 512);
#pragma unroll
      for (int m = 0; m < 4; ++m) {
        v16h a = load_a_frag(hA + m * 16 * 512 + kt * 32, 512);
        acc[0][m] = wmma_f16(a, b0, acc[0][m]);
        acc[1][m] = wmma_f16(a, b1, acc[1][m]);
      }
    }
#pragma unroll
    for (int m = 0; m < 4; ++m) {
      store_c_rm_f(scr + m * 16 * 256 + (nt0 + 0) * 16, 256, acc[0][m]);
      store_c_rm_f(scr + m * 16 * 256 + (nt0 + 1) * 16, 256, acc[1][m]);
    }
  }
  __syncthreads();

  // ---- Phase 8: LN2 + residual -> out (windowed scatter) ------------------
  {
    const int t = tid >> 2, qd = tid & 3;
    const float4* r4 = (const float4*)(scr + t * 256 + qd * 64);
    float s = 0.0f, s2 = 0.0f;
#pragma unroll
    for (int i = 0; i < 16; ++i) {
      float4 v = r4[i];
      s  += v.x + v.y + v.z + v.w;
      s2 += v.x * v.x + v.y * v.y + v.z * v.z + v.w * v.w;
    }
    s  += __shfl_xor(s, 1);  s  += __shfl_xor(s, 2);
    s2 += __shfl_xor(s2, 1); s2 += __shfl_xor(s2, 2);
    float mean = s * (1.0f / 256.0f);
    float var  = s2 * (1.0f / 256.0f) - mean * mean;
    float inv  = rsqrtf(var + 1e-5f);
    const int n = (hy * 8 + (t >> 3)) * GRID_W + wx * 8 + (t & 7);
    const float4* xr4 = (const float4*)(x + xbase + (long)n * C_DIM + qd * 64);
    float4* o4        = (float4*)(out + xbase + (long)n * C_DIM + qd * 64);
    const float4* gg4 = (const float4*)(g2 + qd * 64);
    const float4* bb4 = (const float4*)(b2 + qd * 64);
#pragma unroll
    for (int i = 0; i < 16; ++i) {
      float4 v = r4[i], g = gg4[i], b = bb4[i], xr = xr4[i];
      float4 o;
      o.x = (v.x - mean) * inv * g.x + b.x + xr.x;
      o.y = (v.y - mean) * inv * g.y + b.y + xr.y;
      o.z = (v.z - mean) * inv * g.z + b.z + xr.z;
      o.w = (v.w - mean) * inv * g.w + b.w + xr.w;
      o4[i] = o;
    }
  }
}

// ---------------------------------------------------------------------------
extern "C" void kernel_launch(void* const* d_in, const int* in_sizes, int n_in,
                              void* d_out, int out_size, void* d_ws, size_t ws_size,
                              hipStream_t stream) {
  (void)in_sizes; (void)n_in; (void)out_size; (void)ws_size;
  const float* x   = (const float*)d_in[0];
  const float* Wq  = (const float*)d_in[1];
  const float* Wk  = (const float*)d_in[2];
  const float* Wv  = (const float*)d_in[3];
  const float* Wm  = (const float*)d_in[4];
  const float* W1  = (const float*)d_in[5];
  const float* W2  = (const float*)d_in[6];
  const float* g1  = (const float*)d_in[7];
  const float* b1  = (const float*)d_in[8];
  const float* g2  = (const float*)d_in[9];
  const float* b2  = (const float*)d_in[10];
  float* out = (float*)d_out;

  _Float16* ws  = (_Float16*)d_ws;
  _Float16* pWq = ws;
  _Float16* pWk = ws + 65536;
  _Float16* pWv = ws + 131072;
  _Float16* pWm = ws + 196608;
  _Float16* pW1 = ws + 262144;
  _Float16* pW2 = ws + 524288;

  pack_w_kernel<<<256, 256, 0, stream>>>(Wq, pWq, 256, 256);
  pack_w_kernel<<<256, 256, 0, stream>>>(Wk, pWk, 256, 256);
  pack_w_kernel<<<256, 256, 0, stream>>>(Wv, pWv, 256, 256);
  pack_w_kernel<<<256, 256, 0, stream>>>(Wm, pWm, 256, 256);
  pack_w_kernel<<<1024, 256, 0, stream>>>(W1, pW1, 512, 512);
  pack_w_kernel<<<512, 256, 0, stream>>>(W2, pW2, 512, 256);

  (void)hipFuncSetAttribute(reinterpret_cast<const void*>(loftr_layer_kernel),
                            hipFuncAttributeMaxDynamicSharedMemorySize, SMEM_BYTES);
  loftr_layer_kernel<<<3600, 256, SMEM_BYTES, stream>>>(
      x, pWq, pWk, pWv, pWm, pW1, pW2, g1, b1, g2, b2, out);
}
